// AbsWordCopySummarizer_2010044694962
// MI455X (gfx1250) — matrix operand order repaired and checked by reading, not compiled, not run on hardware
//
#include <hip/hip_runtime.h>
#include <hip/hip_bf16.h>
#include <math.h>

// ---------------- problem constants ----------------
constexpr int B  = 8;
constexpr int HH = 8;      // attention heads
constexpr int T  = 128;    // target length
constexpr int S  = 512;    // source length
constexpr int H  = 768;    // hidden
constexpr int V  = 30522;  // vocab
constexpr int MT = B * T;  // 1024 flattened rows

// ---------------- WMMA / TDM types (gfx1250, wave32) ----------------
typedef __attribute__((ext_vector_type(16))) __bf16       v16bf;
typedef __attribute__((ext_vector_type(8)))  float        v8f;
typedef __attribute__((ext_vector_type(4)))  unsigned int u32x4;
typedef __attribute__((ext_vector_type(4)))  int          i32x4;
typedef __attribute__((ext_vector_type(8)))  int          i32x8;

union Frag16 { u32x4 q[2]; v16bf v; };

__device__ __forceinline__ unsigned pack_bf16(float x, float y) {
    union { __bf16 b[2]; unsigned u; } c;
    c.b[0] = (__bf16)x; c.b[1] = (__bf16)y;
    return c.u;
}
__device__ __forceinline__ __bf16 f2bf(float x) { return (__bf16)x; }

#if __has_builtin(__builtin_amdgcn_tensor_load_to_lds) && \
    __has_builtin(__builtin_amdgcn_s_wait_tensorcnt)
#define HAVE_TDM 1
#else
#define HAVE_TDM 0
#endif

#if HAVE_TDM
// One 2D TDM load: W_gen[n0 .. n0+127][k0 .. k0+31] (f32) -> LDS (packed 128x32 f32).
// tensor_dim1 = V-n0 => rows past the vocab end are zero-filled by the TDM.
__device__ __forceinline__ void tdm_issue(const float* Wg, int n0, int k0,
                                          unsigned lds_off, unsigned rows) {
    unsigned long long ga = (unsigned long long)(size_t)Wg +
                            ((unsigned long long)(unsigned)n0 * H + (unsigned)k0) * 4ull;
    u32x4 g0;
    g0[0] = 1u;                                               // count=1 (valid D#)
    g0[1] = lds_off;                                          // lds_addr (bytes)
    g0[2] = (unsigned)ga;                                     // global_addr[31:0]
    g0[3] = (unsigned)((ga >> 32) & 0x1ffffffull) | (2u << 30); // addr[56:32] | type=2
    i32x8 g1;
    g1[0] = (int)(2u << 16);                                  // wg_mask=0, data_size=4B
    g1[1] = (int)(32u << 16);                                 // tensor_dim0 = 32 (lo16)
    g1[2] = (int)((rows & 0xffffu) << 16);                    // dim0 hi=0 | tensor_dim1 lo16
    g1[3] = (int)(((rows >> 16) & 0xffffu) | (32u << 16));    // dim1 hi16 | tile_dim0=32
    g1[4] = (int)128;                                         // tile_dim1=128, tile_dim2=0
    g1[5] = (int)H;                                           // tensor_dim0_stride = 768
    g1[6] = 0;                                                // stride hi / dim1_stride lo
    g1[7] = 0;
    i32x4 gz = {0, 0, 0, 0};
#if defined(__clang_major__) && __clang_major__ >= 23
    i32x8 gz8 = {0, 0, 0, 0, 0, 0, 0, 0};
    __builtin_amdgcn_tensor_load_to_lds(g0, g1, gz, gz, gz8, 0);
#else
    __builtin_amdgcn_tensor_load_to_lds(g0, g1, gz, gz, 0);
#endif
}
#endif // HAVE_TDM

// =============================================================
// 0) dec [MT,H] f32 -> bf16 (packed pairs) into workspace
// =============================================================
__global__ void cvt_dec_kernel(const float* __restrict__ dec, unsigned* __restrict__ decbf) {
    int i = blockIdx.x * blockDim.x + threadIdx.x;       // over MT*H/2 pairs
    if (i >= MT * H / 2) return;
    float2 f = *(const float2*)(dec + 2 * (size_t)i);
    decbf[i] = pack_bf16(f.x, f.y);
}

// =============================================================
// 1) attn_word[b,t,s] = mean_h attn_dist[b,h,t,s]
// =============================================================
__global__ void mean_heads_kernel(const float* __restrict__ ad, float* __restrict__ aw) {
    int i = blockIdx.x * blockDim.x + threadIdx.x;       // over B*T*S
    if (i >= B * T * S) return;
    int s = i % S;
    int t = (i / S) % T;
    int b = i / (S * T);
    const float* base = ad + (((size_t)b * HH) * T + t) * S + s;
    float sum = 0.f;
#pragma unroll
    for (int h = 0; h < HH; ++h) sum += base[(size_t)h * T * S];
    aw[i] = sum * (1.0f / HH);
}

// =============================================================
// 2) ctx[b,t,h] = sum_s attn_word[b,t,s] * word_vec[b,s,h]  (bf16 WMMA)
// =============================================================
__global__ void ctx_wmma_kernel(const float* __restrict__ aw,
                                const float* __restrict__ wv,
                                float* __restrict__ ctx) {
    const int lane   = threadIdx.x & 31;
    const int wave   = threadIdx.x >> 5;
    const int b      = blockIdx.z;
    const int m0     = blockIdx.y * 16;
    const int n0     = blockIdx.x * 128 + wave * 16;
    const int mA     = m0 + (lane & 15);
    const int nB     = n0 + (lane & 15);
    const int khalfA = (lane >> 4) * 8;
    const int kbaseB = (lane >> 4) * 16;

    const float* arow  = aw + ((size_t)b * T + mA) * S;
    const float* bbase = wv + (size_t)b * S * H;

    v8f acc = {};
    for (int k0 = 0; k0 < S; k0 += 32) {
        v16bf a, bm;
#pragma unroll
        for (int j = 0; j < 8; ++j) {
            int ka = k0 + ((j >> 2) << 4) + khalfA + ((j & 3) << 1);
            float2 av = *(const float2*)(arow + ka);
            a[2 * j]     = f2bf(av.x);
            a[2 * j + 1] = f2bf(av.y);
            int kb = k0 + ((j >> 2) << 4) + khalfA + ((j & 3) << 1);
            bm[2 * j]     = f2bf(bbase[(size_t)kb * H + nB]);
            bm[2 * j + 1] = f2bf(bbase[(size_t)(kb + 1) * H + nB]);
        }
        acc = __builtin_amdgcn_wmma_f32_16x16x32_bf16(false, a, false, bm,
                                                      (short)0, acc, false, false);
    }
    const int col = n0 + (lane & 15);
#pragma unroll
    for (int j = 0; j < 8; ++j) {
        int row = m0 + (lane >> 4) * 8 + j;
        ctx[((size_t)b * T + row) * H + col] = acc[j];
    }
}

// =============================================================
// 3) p[m] = sigmoid( dec[m,:]*Wp[0:H] + ctx[m,:]*Wp[H:2H] + bp )
// =============================================================
__global__ void gate_kernel(const float* __restrict__ dec, const float* __restrict__ ctx,
                            const float* __restrict__ Wp, const float* __restrict__ bp,
                            float* __restrict__ p) {
    int m = blockIdx.x * blockDim.x + threadIdx.x;
    if (m >= MT) return;
    const float* d = dec + (size_t)m * H;
    const float* c = ctx + (size_t)m * H;
    float acc = bp[0];
    for (int h = 0; h < H; ++h) acc = fmaf(d[h], Wp[h], acc);
    for (int h = 0; h < H; ++h) acc = fmaf(c[h], Wp[H + h], acc);
    p[m] = 1.0f / (1.0f + __expf(-acc));
}

// =============================================================
// 4) logits[m,v] = dec[m,:] . W_gen[v,:] + b_gen[v]  (into d_out)
//    Block = 128 rows x 128 cols, 8 waves; W_gen tile staged via TDM,
//    converted once to padded bf16 in LDS, 8 WMMAs per wave per K-step.
// =============================================================
__global__ void __launch_bounds__(256)
logits_tdm_wmma_kernel(const unsigned* __restrict__ decbf,  // [MT,H] bf16 pairs
                       const float* __restrict__ Wg,
                       const float* __restrict__ bg,
                       float* __restrict__ out) {
    constexpr int KSTEP = 32;
    constexpr int ROWU  = 20;                 // uints per padded bf16 row (80 B)
#if HAVE_TDM
    __shared__ float    ldsF[128 * KSTEP];    // 16 KB f32 stage (TDM destination)
#endif
    __shared__ unsigned ldsB[128 * ROWU];     // 10 KB bf16 tile (padded rows)

    const int tid  = threadIdx.x;
    const int wave = tid >> 5;
    const int lane = tid & 31;
    const int ln   = lane & 15;
    const int half = lane >> 4;
    const int n0   = blockIdx.x * 128;
    const int m0   = blockIdx.y * 128;
    const int mA   = m0 + wave * 16 + ln;

    v8f acc[8];
#pragma unroll
    for (int j = 0; j < 8; ++j) acc[j] = (v8f){0, 0, 0, 0, 0, 0, 0, 0};

#if HAVE_TDM
    const unsigned ldsf_off = (unsigned)(size_t)(&ldsF[0]);   // LDS byte offset
    const unsigned rows     = (unsigned)(V - n0);
    if (wave == 0) tdm_issue(Wg, n0, 0, ldsf_off, rows);      // prologue load
#endif

    for (int k0 = 0; k0 < H; k0 += KSTEP) {
#if HAVE_TDM
        if (wave == 0) __builtin_amdgcn_s_wait_tensorcnt((short)0);
#endif
        __syncthreads();   // tile visible to all; prev-iter LDS reads done
        // ---- convert staged f32 tile -> padded bf16 (once per block) ----
#pragma unroll
        for (int i = 0; i < 8; ++i) {
            int mp = tid + 256 * i;               // pair index 0..2047
            int r  = mp >> 4;                     // row within tile (0..127)
            int jp = mp & 15;                     // dword pair within row
#if HAVE_TDM
            float2 f = *(const float2*)(&ldsF[2 * mp]);
#else
            int gr = n0 + r;
            float2 f;
            if (gr < V) f = *(const float2*)(Wg + (size_t)gr * H + k0 + 2 * jp);
            else        f = make_float2(0.f, 0.f);
#endif
            ldsB[r * ROWU + jp] = pack_bf16(f.x, f.y);
        }
        __syncthreads();
#if HAVE_TDM
        if (wave == 0 && (k0 + KSTEP) < H)        // pipeline next tile
            tdm_issue(Wg, n0, k0 + KSTEP, ldsf_off, rows);
#endif
        // ---- A fragment: two 16B loads of L2-hot bf16 ----
        Frag16 a;
        const unsigned* ap = decbf + ((size_t)mA * H + k0) / 2 + half * 4;
        a.q[0] = *(const u32x4*)(ap);
        a.q[1] = *(const u32x4*)(ap + 8);         // +16 bf16
        // ---- 8 WMMAs against the shared B tile ----
#pragma unroll
        for (int j = 0; j < 8; ++j) {
            Frag16 bfr;
            const unsigned* bp = &ldsB[((j << 4) + ln) * ROWU + half * 4];
            bfr.q[0] = *(const u32x4*)(bp);
            bfr.q[1] = *(const u32x4*)(bp + 8);
            acc[j] = __builtin_amdgcn_wmma_f32_16x16x32_bf16(false, a.v, false, bfr.v,
                                                             (short)0, acc[j], false, false);
        }
    }

    // ---- store: bias + logits ----
#pragma unroll
    for (int j = 0; j < 8; ++j) {
        int col = n0 + j * 16 + ln;
        if (col < V) {
            float bias = bg[col];
            size_t rbase = (size_t)(m0 + wave * 16 + half * 8) * V + col;
#pragma unroll
            for (int r = 0; r < 8; ++r)
                out[rbase + (size_t)r * V] = acc[j][r] + bias;
        }
    }
}

// =============================================================
// 5) per-row max and sum(exp(x-max)) over V  (one block per row)
// =============================================================
__global__ void rowstat_kernel(const float* __restrict__ logits,
                               float* __restrict__ rmax, float* __restrict__ rsum) {
    const int m = blockIdx.x;
    const float* row = logits + (size_t)m * V;
    __shared__ float red[256];
    float mx = -INFINITY;
    for (int v = threadIdx.x; v < V; v += 256) mx = fmaxf(mx, row[v]);
    red[threadIdx.x] = mx;
    __syncthreads();
    for (int off = 128; off > 0; off >>= 1) {
        if (threadIdx.x < off) red[threadIdx.x] = fmaxf(red[threadIdx.x], red[threadIdx.x + off]);
        __syncthreads();
    }
    mx = red[0];
    __syncthreads();
    float sum = 0.f;
    for (int v = threadIdx.x; v < V; v += 256) sum += __expf(row[v] - mx);
    red[threadIdx.x] = sum;
    __syncthreads();
    for (int off = 128; off > 0; off >>= 1) {
        if (threadIdx.x < off) red[threadIdx.x] += red[threadIdx.x + off];
        __syncthreads();
    }
    if (threadIdx.x == 0) { rmax[m] = mx; rsum[m] = red[0]; }
}

// =============================================================
// 6) in place: out[m,v] = p[m] * exp(out[m,v]-rmax[m]) / rsum[m]
// =============================================================
__global__ void finalize_kernel(float* __restrict__ out,
                                const float* __restrict__ rmax,
                                const float* __restrict__ rsum,
                                const float* __restrict__ p) {
    size_t i = (size_t)blockIdx.x * blockDim.x + threadIdx.x;
    if (i >= (size_t)MT * V) return;
    int m = (int)(i / V);
    float g = __expf(out[i] - rmax[m]) / rsum[m];
    out[i] = p[m] * g;
}

// =============================================================
// 7) out[m, words[b,s]] += (1-p[m]) * attn_word[b,t,s]
// =============================================================
__global__ void scatter_kernel(float* __restrict__ out,
                               const float* __restrict__ aw,
                               const int* __restrict__ words,
                               const float* __restrict__ p) {
    int i = blockIdx.x * blockDim.x + threadIdx.x;   // over B*T*S
    if (i >= B * T * S) return;
    int s = i % S;
    int m = i / S;
    int b = m / T;
    int w = words[b * S + s];
    atomicAdd(out + (size_t)m * V + w, (1.0f - p[m]) * aw[i]);
}

// =============================================================
extern "C" void kernel_launch(void* const* d_in, const int* in_sizes, int n_in,
                              void* d_out, int out_size, void* d_ws, size_t ws_size,
                              hipStream_t stream) {
    (void)in_sizes; (void)n_in; (void)out_size; (void)ws_size;

    const float* dec   = (const float*)d_in[0];   // [B,T,H]
    const float* ad    = (const float*)d_in[1];   // [B,HH,T,S]
    const float* wv    = (const float*)d_in[2];   // [B,S,H]
    const int*   words = (const int*)  d_in[3];   // [B,S]
    const float* Wg    = (const float*)d_in[4];   // [V,H]
    const float* bg    = (const float*)d_in[5];   // [V]
    const float* Wp    = (const float*)d_in[6];   // [1,2H]
    const float* bp    = (const float*)d_in[7];   // [1]
    float* out = (float*)d_out;                   // [B,T,V]

    // workspace layout: ~6.8 MB total
    float*    aw    = (float*)d_ws;                    // B*T*S   = 524288 f32
    float*    ctx   = aw  + (size_t)B * T * S;         // MT*H    = 786432 f32
    float*    p     = ctx + (size_t)MT * H;            // 1024 f32
    float*    rmax  = p    + MT;                       // 1024 f32
    float*    rsum  = rmax + MT;                       // 1024 f32
    unsigned* decbf = (unsigned*)(rsum + MT);          // MT*H/2 = 393216 u32 (bf16 pairs)

    // 0) dec -> bf16
    cvt_dec_kernel<<<(MT * H / 2 + 255) / 256, 256, 0, stream>>>(dec, decbf);
    // 1) head mean
    mean_heads_kernel<<<(B * T * S + 255) / 256, 256, 0, stream>>>(ad, aw);
    // 2) attention context (WMMA)
    ctx_wmma_kernel<<<dim3(H / 128, T / 16, B), 256, 0, stream>>>(aw, wv, ctx);
    // 3) copy gate
    gate_kernel<<<(MT + 255) / 256, 256, 0, stream>>>(dec, ctx, Wp, bp, p);
    // 4) generator logits into d_out (TDM-staged bf16 WMMA)
    logits_tdm_wmma_kernel<<<dim3((V + 127) / 128, MT / 128), 256, 0, stream>>>(decbf, Wg, bg, out);
    // 5) softmax row stats
    rowstat_kernel<<<MT, 256, 0, stream>>>(out, rmax, rsum);
    // 6) out = p * softmax(logits)
    size_t total = (size_t)MT * V;
    finalize_kernel<<<(unsigned)((total + 255) / 256), 256, 0, stream>>>(out, rmax, rsum, p);
    // 7) copy-distribution scatter
    scatter_kernel<<<(B * T * S + 255) / 256, 256, 0, stream>>>(out, aw, words, p);
}
